// TransitionPathDiffusionGNN_63093069578775
// MI455X (gfx1250) — matrix operand order repaired and checked by reading, not compiled, not run on hardware
//
#include <hip/hip_runtime.h>
#include <hip/hip_bf16.h>
#include <math.h>

// ---------------------------------------------------------------------------
// TransitionPathDiffusionGNN on MI455X (gfx1250, wave32, WMMA f16->f32)
// Edge MLPs: 64-edge M-tiles, 4x B-fragment reuse per wave.
// ---------------------------------------------------------------------------

typedef __attribute__((ext_vector_type(16))) _Float16 v16h;
typedef __attribute__((ext_vector_type(8)))  _Float16 v8h;
typedef __attribute__((ext_vector_type(8)))  float    v8f;

#define NN        8192
#define EE        131072
#define STATE     352
#define MSGD      128
#define HIDD      128
#define NRBF      10
#define KEDGE     768   // 2*STATE + 37 = 741 padded to multiple of 32
#define KEDGE_REAL 741
#define KSTATE    480   // STATE + MSG (already multiple of 32)
#define EPB       64    // edges per block (4 M-subtiles of 16)

__device__ __forceinline__ float gelu_f(float x) {
  const float k0 = 0.7978845608028654f;   // sqrt(2/pi)
  const float k1 = 0.044715f;
  return 0.5f * x * (1.0f + tanhf(k0 * (x + k1 * x * x * x)));
}

// ---- single 16x16 tile (node kernels) -------------------------------------
__device__ __forceinline__ v8f wmma_tile_f16(const _Float16* __restrict__ A, int lda,
                                             const _Float16* __restrict__ Bt, int ldb,
                                             int K, v8f acc)
{
  const int lane = threadIdx.x & 31;
  const int hl   = lane >> 4;
  const _Float16* arow = A  + (lane & 15) * lda;
  const _Float16* bcol = Bt + (lane & 15) * ldb;
  for (int k0 = 0; k0 < K; k0 += 32) {
    union { v16h v; v8h h[2]; } a, b;
    a.h[0] = *(const v8h*)(arow + k0 + hl * 8);
    a.h[1] = *(const v8h*)(arow + k0 + 16 + hl * 8);
    b.h[0] = *(const v8h*)(bcol + k0 + hl * 8);
    b.h[1] = *(const v8h*)(bcol + k0 + 16 + hl * 8);
    acc = __builtin_amdgcn_wmma_f32_16x16x32_f16(false, a.v, false, b.v,
                                                 (short)0, acc, false, false);
  }
  return acc;
}

// ---- 64x16 tile: one B fragment feeds 4 WMMAs (edge kernels) --------------
__device__ __forceinline__ void wmma_mtile4(const _Float16* __restrict__ A, int lda,
                                            const _Float16* __restrict__ Bt, int ldb,
                                            int K, v8f acc[4])
{
  const int lane = threadIdx.x & 31;
  const int hl   = lane >> 4;
  const _Float16* arow = A  + (lane & 15) * lda;
  const _Float16* bcol = Bt + (lane & 15) * ldb;
  for (int k0 = 0; k0 < K; k0 += 32) {
    union { v16h v; v8h h[2]; } b;
    b.h[0] = *(const v8h*)(bcol + k0 + hl * 8);
    b.h[1] = *(const v8h*)(bcol + k0 + 16 + hl * 8);
#pragma unroll
    for (int mt = 0; mt < 4; ++mt) {
      union { v16h v; v8h h[2]; } a;
      const _Float16* ar = arow + mt * 16 * lda;
      a.h[0] = *(const v8h*)(ar + k0 + hl * 8);
      a.h[1] = *(const v8h*)(ar + k0 + 16 + hl * 8);
      acc[mt] = __builtin_amdgcn_wmma_f32_16x16x32_f16(false, a.v, false, b.v,
                                                       (short)0, acc[mt], false, false);
    }
  }
}

__device__ __forceinline__ void store_tile_gelu(v8f acc, const float* __restrict__ bias,
                                                int ncol0, _Float16* __restrict__ out, int ldo)
{
  const int lane = threadIdx.x & 31;
  const int hl   = lane >> 4;
  const int n    = ncol0 + (lane & 15);
  const float bv = bias[n];
#pragma unroll
  for (int i = 0; i < 8; ++i)
    out[(i + 8 * hl) * ldo + n] = (_Float16)gelu_f(acc[i] + bv);
}

__device__ __forceinline__ void store_mtile_gelu(v8f acc[4], const float* __restrict__ bias,
                                                 int ncol0, _Float16* __restrict__ out, int ldo)
{
  const int lane = threadIdx.x & 31;
  const int hl   = lane >> 4;
  const int n    = ncol0 + (lane & 15);
  const float bv = bias[n];
#pragma unroll
  for (int mt = 0; mt < 4; ++mt)
#pragma unroll
    for (int i = 0; i < 8; ++i)
      out[(mt * 16 + i + 8 * hl) * ldo + n] = (_Float16)gelu_f(acc[mt][i] + bv);
}

// 37 edge features -> f16 row tail; also emits dx for the alpha kernel.
__device__ __forceinline__ void fill_edge_features(
    int e, int sN, int dN,
    const float* __restrict__ x,  const float* __restrict__ xA, const float* __restrict__ xB,
    const float* __restrict__ bondA, const float* __restrict__ bondB,
    _Float16* __restrict__ f, float* __restrict__ dxo)
{
  const float eps = 1e-12f;
  float dx0 = x[sN*3+0]-x[dN*3+0], dx1 = x[sN*3+1]-x[dN*3+1], dx2 = x[sN*3+2]-x[dN*3+2];
  dxo[0]=dx0; dxo[1]=dx1; dxo[2]=dx2;
  float d2   = dx0*dx0 + dx1*dx1 + dx2*dx2 + eps;
  float dist = sqrtf(d2);
  float a0 = xA[sN*3+0]-xA[dN*3+0], a1 = xA[sN*3+1]-xA[dN*3+1], a2 = xA[sN*3+2]-xA[dN*3+2];
  float dA = sqrtf(a0*a0 + a1*a1 + a2*a2 + eps);
  float b0 = xB[sN*3+0]-xB[dN*3+0], b1 = xB[sN*3+1]-xB[dN*3+1], b2 = xB[sN*3+2]-xB[dN*3+2];
  float dB = sqrtf(b0*b0 + b1*b1 + b2*b2 + eps);
  f[0] = (_Float16)bondA[e]; f[1] = (_Float16)bondB[e];
  f[2] = (_Float16)dist;     f[3] = (_Float16)d2;
  f[4] = (_Float16)dA;       f[5] = (_Float16)dB;  f[6] = (_Float16)(dA - dB);
  const float delta = 5.0f / 9.0f, invd = 9.0f / 5.0f;
#pragma unroll
  for (int j = 0; j < NRBF; ++j) {
    float mu = delta * (float)j;
    float u0 = (dist - mu) * invd, uA = (dA - mu) * invd, uB = (dB - mu) * invd;
    f[7 + j]          = (_Float16)expf(-u0 * u0);
    f[7 + NRBF + j]   = (_Float16)expf(-uA * uA);
    f[7 + 2*NRBF + j] = (_Float16)expf(-uB * uB);
  }
#pragma unroll
  for (int j = 37; j < KEDGE - 2 * STATE; ++j) f[j] = (_Float16)0.0f;   // zero pad 741..767
}

// ---------------------------------------------------------------------------
// Weight conversion: fp32 W[Kin][Nout] -> f16 transposed padded Wt[Nout][Kpad]
// ---------------------------------------------------------------------------
__global__ void convert_weight_kernel(const float* __restrict__ W, _Float16* __restrict__ Wt,
                                      int Kin, int Kpad, int Nout)
{
  int i = blockIdx.x * 256 + threadIdx.x;
  if (i >= Nout * Kpad) return;
  int n = i / Kpad, k = i % Kpad;
  Wt[i] = (k < Kin) ? (_Float16)W[k * Nout + n] : (_Float16)0.0f;
}

// ---------------------------------------------------------------------------
// Node init: atom_info MLP (20->64 gelu ->64) + fourier(s,t); builds h (f32+f16)
// ---------------------------------------------------------------------------
__global__ void node_init_kernel(const float* __restrict__ Z, const float* __restrict__ hA,
                                 const float* __restrict__ hB, const float* __restrict__ s,
                                 const float* __restrict__ t,
                                 const float* __restrict__ W0, const float* __restrict__ B0,
                                 const float* __restrict__ W1, const float* __restrict__ B1,
                                 const float* __restrict__ x_t, float* __restrict__ xcur,
                                 float* __restrict__ hF, _Float16* __restrict__ h16)
{
  int n = blockIdx.x * 128 + threadIdx.x;
  if (n >= NN) return;
  float hid[64];
  for (int j = 0; j < 64; ++j) {
    float acc = B0[j];
    for (int k = 0; k < 20; ++k) acc += Z[n * 20 + k] * W0[k * 64 + j];
    hid[j] = gelu_f(acc);
  }
  float*     hr  = hF  + (size_t)n * STATE;
  _Float16*  hr16= h16 + (size_t)n * STATE;
  for (int j = 0; j < 64; ++j) {
    float acc = B1[j];
    for (int k = 0; k < 64; ++k) acc += hid[k] * W1[k * 64 + j];
    hr[j] = acc; hr16[j] = (_Float16)acc;
  }
  for (int j = 0; j < 128; ++j) { float v = hA[n*128+j]; hr[64+j]  = v; hr16[64+j]  = (_Float16)v; }
  for (int j = 0; j < 128; ++j) { float v = hB[n*128+j]; hr[192+j] = v; hr16[192+j] = (_Float16)v; }
  const float pi = 3.14159265358979f;
  float sv = s[n], tv = t[n];
  for (int k = 0; k < 8; ++k) {
    float as = pi * sv * (float)(k + 1), at = pi * tv * (float)(k + 1);
    float v;
    v = sinf(as); hr[320+k] = v; hr16[320+k] = (_Float16)v;
    v = cosf(as); hr[328+k] = v; hr16[328+k] = (_Float16)v;
    v = sinf(at); hr[336+k] = v; hr16[336+k] = (_Float16)v;
    v = cosf(at); hr[344+k] = v; hr16[344+k] = (_Float16)v;
  }
  xcur[n*3+0] = x_t[n*3+0]; xcur[n*3+1] = x_t[n*3+1]; xcur[n*3+2] = x_t[n*3+2];
}

// ---------------------------------------------------------------------------
// Edge message MLP (741->128->128->128) + segment_sum via atomics. 64 edges/WG.
// ---------------------------------------------------------------------------
__global__ void __launch_bounds__(256)
edge_message_kernel(const _Float16* __restrict__ h16, const float* __restrict__ x,
                    const float* __restrict__ xA, const float* __restrict__ xB,
                    const float* __restrict__ bondA, const float* __restrict__ bondB,
                    const int* __restrict__ src, const int* __restrict__ dst,
                    const _Float16* __restrict__ W1t, const float* __restrict__ b1,
                    const _Float16* __restrict__ W2t, const float* __restrict__ b2,
                    const _Float16* __restrict__ W3t, const float* __restrict__ b3,
                    float* __restrict__ node_msg)
{
  __shared__ alignas(32) _Float16 Ain[EPB][KEDGE];    // 96 KB
  __shared__ alignas(32) _Float16 actA[EPB][HIDD];    // 16 KB
  __shared__ alignas(32) _Float16 actB[EPB][HIDD];    // 16 KB
  __shared__ int   sSrc[EPB], sDst[EPB];
  __shared__ float sdx[EPB][3];
  const int tid = threadIdx.x;
  const int e0  = blockIdx.x * EPB;

  if (tid < EPB) { sSrc[tid] = src[e0 + tid]; sDst[tid] = dst[e0 + tid]; }
  __syncthreads();
  // Gather h[src] | h[dst] (L2-resident 5.8 MB table) into the A tile
  for (int it = tid; it < EPB * (STATE / 8); it += 256) {
    const int r = it / 44, c = (it % 44) * 8;
    const int sN = sSrc[r], dN = sDst[r];
    *(v8h*)&Ain[r][c]         = *(const v8h*)&h16[(size_t)sN * STATE + c];
    *(v8h*)&Ain[r][STATE + c] = *(const v8h*)&h16[(size_t)dN * STATE + c];
  }
  if (tid < EPB)
    fill_edge_features(e0 + tid, sSrc[tid], sDst[tid], x, xA, xB, bondA, bondB,
                       &Ain[tid][2 * STATE], sdx[tid]);
  __syncthreads();

  const int wave = tid >> 5, lane = tid & 31, hl = lane >> 4;
  v8f acc1[4] = {};
  wmma_mtile4(&Ain[0][0], KEDGE, W1t + wave * 16 * KEDGE, KEDGE, KEDGE, acc1);
  store_mtile_gelu(acc1, b1, wave * 16, &actA[0][0], HIDD);
  __syncthreads();
  v8f acc2[4] = {};
  wmma_mtile4(&actA[0][0], HIDD, W2t + wave * 16 * HIDD, HIDD, HIDD, acc2);
  store_mtile_gelu(acc2, b2, wave * 16, &actB[0][0], HIDD);
  __syncthreads();
  v8f acc3[4] = {};
  wmma_mtile4(&actB[0][0], HIDD, W3t + wave * 16 * HIDD, HIDD, HIDD, acc3);
  const int n  = wave * 16 + (lane & 15);
  const float bv = b3[n];
#pragma unroll
  for (int mt = 0; mt < 4; ++mt)
#pragma unroll
    for (int i = 0; i < 8; ++i) {
      const int r = mt * 16 + i + 8 * hl;
      atomicAdd(&node_msg[(size_t)sDst[r] * MSGD + n], acc3[mt][i] + bv);
    }
}

// ---------------------------------------------------------------------------
// Edge alpha MLP (741->128->128->1) + neighbor_update += alpha*dx (atomics)
// ---------------------------------------------------------------------------
__global__ void __launch_bounds__(256)
edge_alpha_kernel(const _Float16* __restrict__ h16, const float* __restrict__ x,
                  const float* __restrict__ xA, const float* __restrict__ xB,
                  const float* __restrict__ bondA, const float* __restrict__ bondB,
                  const int* __restrict__ src, const int* __restrict__ dst,
                  const _Float16* __restrict__ W1t, const float* __restrict__ b1,
                  const _Float16* __restrict__ W2t, const float* __restrict__ b2,
                  const float* __restrict__ W3, const float* __restrict__ b3,
                  float* __restrict__ upd)
{
  __shared__ alignas(32) _Float16 Ain[EPB][KEDGE];
  __shared__ alignas(32) _Float16 actA[EPB][HIDD];
  __shared__ alignas(32) _Float16 actB[EPB][HIDD];
  __shared__ int   sSrc[EPB], sDst[EPB];
  __shared__ float sdx[EPB][3];
  const int tid = threadIdx.x;
  const int e0  = blockIdx.x * EPB;

  if (tid < EPB) { sSrc[tid] = src[e0 + tid]; sDst[tid] = dst[e0 + tid]; }
  __syncthreads();
  for (int it = tid; it < EPB * (STATE / 8); it += 256) {
    const int r = it / 44, c = (it % 44) * 8;
    const int sN = sSrc[r], dN = sDst[r];
    *(v8h*)&Ain[r][c]         = *(const v8h*)&h16[(size_t)sN * STATE + c];
    *(v8h*)&Ain[r][STATE + c] = *(const v8h*)&h16[(size_t)dN * STATE + c];
  }
  if (tid < EPB)
    fill_edge_features(e0 + tid, sSrc[tid], sDst[tid], x, xA, xB, bondA, bondB,
                       &Ain[tid][2 * STATE], sdx[tid]);
  __syncthreads();

  const int wave = tid >> 5;
  v8f acc1[4] = {};
  wmma_mtile4(&Ain[0][0], KEDGE, W1t + wave * 16 * KEDGE, KEDGE, KEDGE, acc1);
  store_mtile_gelu(acc1, b1, wave * 16, &actA[0][0], HIDD);
  __syncthreads();
  v8f acc2[4] = {};
  wmma_mtile4(&actA[0][0], HIDD, W2t + wave * 16 * HIDD, HIDD, HIDD, acc2);
  store_mtile_gelu(acc2, b2, wave * 16, &actB[0][0], HIDD);
  __syncthreads();
  if (tid < EPB) {
    float acc = b3[0];
    for (int k = 0; k < HIDD; ++k) acc += (float)actB[tid][k] * W3[k];
    const int dN = sDst[tid];
    atomicAdd(&upd[dN * 3 + 0], acc * sdx[tid][0]);
    atomicAdd(&upd[dN * 3 + 1], acc * sdx[tid][1]);
    atomicAdd(&upd[dN * 3 + 2], acc * sdx[tid][2]);
  }
}

// ---------------------------------------------------------------------------
// State MLP (480->128->128->352) with residual, refresh f32+f16 h. 16 nodes/WG.
// ---------------------------------------------------------------------------
__global__ void __launch_bounds__(256)
node_state_kernel(float* __restrict__ hF, _Float16* __restrict__ h16,
                  const float* __restrict__ node_msg,
                  const _Float16* __restrict__ W1t, const float* __restrict__ b1,
                  const _Float16* __restrict__ W2t, const float* __restrict__ b2,
                  const _Float16* __restrict__ W3t, const float* __restrict__ b3)
{
  __shared__ alignas(32) _Float16 Ain[16][KSTATE];
  __shared__ alignas(32) _Float16 actA[16][HIDD];
  __shared__ alignas(32) _Float16 actB[16][HIDD];
  const int tid = threadIdx.x;
  const int n0  = blockIdx.x * 16;

  for (int it = tid; it < 16 * (STATE / 8); it += 256) {
    const int r = it / 44, c = (it % 44) * 8;
    *(v8h*)&Ain[r][c] = *(const v8h*)&h16[(size_t)(n0 + r) * STATE + c];
  }
  for (int it = tid; it < 16 * MSGD; it += 256) {
    const int r = it / MSGD, c = it % MSGD;
    Ain[r][STATE + c] = (_Float16)node_msg[(size_t)(n0 + r) * MSGD + c];
  }
  __syncthreads();

  const int wave = tid >> 5, lane = tid & 31, hl = lane >> 4;
  v8f z = {};
  v8f a1 = wmma_tile_f16(&Ain[0][0], KSTATE, W1t + wave * 16 * KSTATE, KSTATE, KSTATE, z);
  store_tile_gelu(a1, b1, wave * 16, &actA[0][0], HIDD);
  __syncthreads();
  v8f a2 = wmma_tile_f16(&actA[0][0], HIDD, W2t + wave * 16 * HIDD, HIDD, HIDD, z);
  store_tile_gelu(a2, b2, wave * 16, &actB[0][0], HIDD);
  __syncthreads();
  for (int nt = wave; nt < STATE / 16; nt += 8) {          // 22 output col tiles
    v8f a3 = wmma_tile_f16(&actB[0][0], HIDD, W3t + nt * 16 * HIDD, HIDD, HIDD, z);
    const int n = nt * 16 + (lane & 15);
    const float bv = b3[n];
#pragma unroll
    for (int i = 0; i < 8; ++i) {
      const int r = i + 8 * hl;
      const size_t idx = (size_t)(n0 + r) * STATE + n;
      const float hn = hF[idx] + a3[i] + bv;               // residual
      hF[idx]  = hn;
      h16[idx] = (_Float16)hn;
    }
  }
}

// ---------------------------------------------------------------------------
// Scalar-head node MLP (352->128->128->1) for beta/gamma. 16 nodes/WG.
// ---------------------------------------------------------------------------
__global__ void __launch_bounds__(256)
node_scalar_mlp_kernel(const _Float16* __restrict__ h16,
                       const _Float16* __restrict__ W1t, const float* __restrict__ b1,
                       const _Float16* __restrict__ W2t, const float* __restrict__ b2,
                       const float* __restrict__ W3, const float* __restrict__ b3,
                       float* __restrict__ out)
{
  __shared__ alignas(32) _Float16 actA[16][HIDD];
  __shared__ alignas(32) _Float16 actB[16][HIDD];
  const int tid = threadIdx.x;
  const int n0  = blockIdx.x * 16;
  const int wave = tid >> 5;
  const _Float16* A = h16 + (size_t)n0 * STATE;            // A rows straight from global
  v8f z = {};
  v8f a1 = wmma_tile_f16(A, STATE, W1t + wave * 16 * STATE, STATE, STATE, z);
  store_tile_gelu(a1, b1, wave * 16, &actA[0][0], HIDD);
  __syncthreads();
  v8f a2 = wmma_tile_f16(&actA[0][0], HIDD, W2t + wave * 16 * HIDD, HIDD, HIDD, z);
  store_tile_gelu(a2, b2, wave * 16, &actB[0][0], HIDD);
  __syncthreads();
  if (tid < 16) {
    float acc = b3[0];
    for (int k = 0; k < HIDD; ++k) acc += (float)actB[tid][k] * W3[k];
    out[n0 + tid] = acc;
  }
}

// ---------------------------------------------------------------------------
// Position update & recenter
// ---------------------------------------------------------------------------
__global__ void x_update_kernel(float* __restrict__ x, const float* __restrict__ upd,
                                const float* __restrict__ beta, const float* __restrict__ gmm,
                                const float* __restrict__ s,
                                const float* __restrict__ xA, const float* __restrict__ xB)
{
  int n = blockIdx.x * 256 + threadIdx.x;
  if (n >= NN) return;
  float be = beta[n] * (1.0f - s[n]);
  float ga = gmm[n] * s[n];
#pragma unroll
  for (int c = 0; c < 3; ++c) {
    float xv = x[n * 3 + c];
    x[n * 3 + c] = xv + upd[n * 3 + c] + be * (xA[n * 3 + c] - xv) + ga * (xB[n * 3 + c] - xv);
  }
}

__global__ void reduce_sum_kernel(const float* __restrict__ x, float* __restrict__ sums)
{
  __shared__ float sh[256][3];
  const int tid = threadIdx.x;
  const int n = blockIdx.x * 256 + tid;
#pragma unroll
  for (int c = 0; c < 3; ++c) sh[tid][c] = (n < NN) ? x[n * 3 + c] : 0.0f;
  __syncthreads();
  for (int st = 128; st > 0; st >>= 1) {
    if (tid < st) {
#pragma unroll
      for (int c = 0; c < 3; ++c) sh[tid][c] += sh[tid + st][c];
    }
    __syncthreads();
  }
  if (tid == 0) {
    atomicAdd(&sums[0], sh[0][0]);
    atomicAdd(&sums[1], sh[0][1]);
    atomicAdd(&sums[2], sh[0][2]);
  }
}

__global__ void recenter_kernel(const float* __restrict__ x, const float* __restrict__ sums,
                                float* __restrict__ out)
{
  int i = blockIdx.x * 256 + threadIdx.x;
  if (i < NN * 3) out[i] = x[i] - sums[i % 3] * (1.0f / (float)NN);
}

// ---------------------------------------------------------------------------
// Host orchestration
// ---------------------------------------------------------------------------
extern "C" void kernel_launch(void* const* d_in, const int* in_sizes, int n_in,
                              void* d_out, int out_size, void* d_ws, size_t ws_size,
                              hipStream_t stream)
{
  (void)in_sizes; (void)n_in; (void)out_size; (void)ws_size;
  const float* x_t   = (const float*)d_in[0];
  const float* xA    = (const float*)d_in[1];
  const float* xB    = (const float*)d_in[2];
  const float* s_in  = (const float*)d_in[3];
  const float* t_in  = (const float*)d_in[4];
  const float* Zinfo = (const float*)d_in[5];
  const float* hA    = (const float*)d_in[6];
  const float* hB    = (const float*)d_in[7];
  const float* bondA = (const float*)d_in[8];
  const float* bondB = (const float*)d_in[9];
  const int*   src   = (const int*)d_in[10];
  const int*   dst   = (const int*)d_in[11];

  // params flattened in jax pytree order (dict keys sorted):
  //   alpha[2]{W,b}x3  atom_info{W,b}x2  beta[2]{W,b}x3  gamma[2]{W,b}x3
  //   message[2]{W,b}x3  state[2]{W,b}x3   (tuple order W then b)
  const int P = 12;
  auto pw = [&](int rel) { return (const float*)d_in[P + rel]; };
  const int OFF_ALPHA = 0, OFF_ATOM = 12, OFF_BETA = 16, OFF_GAMMA = 28,
            OFF_MSG = 40, OFF_STATE = 52;

  // ---- workspace carve (256B aligned) ----
  char* ws = (char*)d_ws;
  size_t off = 0;
  auto carve = [&](size_t bytes) -> char* {
    char* p = ws + off;
    off = (off + bytes + 255) & ~(size_t)255;
    return p;
  };
  float*    hF   = (float*)   carve((size_t)NN * STATE * 4);
  _Float16* h16  = (_Float16*)carve((size_t)NN * STATE * 2);
  float*    xcur = (float*)   carve((size_t)NN * 3 * 4);
  float*    upd  = (float*)   carve((size_t)NN * 3 * 4);
  float*    nmsg = (float*)   carve((size_t)NN * MSGD * 4);
  float*    beta = (float*)   carve((size_t)NN * 4);
  float*    gmm  = (float*)   carve((size_t)NN * 4);
  float*    sums = (float*)   carve(16);

  auto convW = [&](const float* W, int Kin, int Kpad, int Nout) -> _Float16* {
    _Float16* p = (_Float16*)carve((size_t)Nout * Kpad * 2);
    int total = Nout * Kpad;
    convert_weight_kernel<<<(total + 255) / 256, 256, 0, stream>>>(W, p, Kin, Kpad, Nout);
    return p;
  };

  _Float16 *mW1[2], *mW2[2], *mW3[2], *aW1[2], *aW2[2];
  _Float16 *sW1[2], *sW2[2], *sW3[2], *bW1[2], *bW2[2], *gW1[2], *gW2[2];
  for (int l = 0; l < 2; ++l) {
    mW1[l] = convW(pw(OFF_MSG   + 6*l + 0), KEDGE_REAL, KEDGE, 128);
    mW2[l] = convW(pw(OFF_MSG   + 6*l + 2), 128, 128, 128);
    mW3[l] = convW(pw(OFF_MSG   + 6*l + 4), 128, 128, 128);
    aW1[l] = convW(pw(OFF_ALPHA + 6*l + 0), KEDGE_REAL, KEDGE, 128);
    aW2[l] = convW(pw(OFF_ALPHA + 6*l + 2), 128, 128, 128);
    sW1[l] = convW(pw(OFF_STATE + 6*l + 0), KSTATE, KSTATE, 128);
    sW2[l] = convW(pw(OFF_STATE + 6*l + 2), 128, 128, 128);
    sW3[l] = convW(pw(OFF_STATE + 6*l + 4), 128, 128, STATE);
    bW1[l] = convW(pw(OFF_BETA  + 6*l + 0), STATE, STATE, 128);
    bW2[l] = convW(pw(OFF_BETA  + 6*l + 2), 128, 128, 128);
    gW1[l] = convW(pw(OFF_GAMMA + 6*l + 0), STATE, STATE, 128);
    gW2[l] = convW(pw(OFF_GAMMA + 6*l + 2), 128, 128, 128);
  }

  node_init_kernel<<<NN / 128, 128, 0, stream>>>(
      Zinfo, hA, hB, s_in, t_in,
      pw(OFF_ATOM + 0), pw(OFF_ATOM + 1), pw(OFF_ATOM + 2), pw(OFF_ATOM + 3),
      x_t, xcur, hF, h16);

  for (int l = 0; l < 2; ++l) {
    hipMemsetAsync(nmsg, 0, (size_t)NN * MSGD * 4, stream);
    edge_message_kernel<<<EE / EPB, 256, 0, stream>>>(
        h16, xcur, xA, xB, bondA, bondB, src, dst,
        mW1[l], pw(OFF_MSG + 6*l + 1), mW2[l], pw(OFF_MSG + 6*l + 3),
        mW3[l], pw(OFF_MSG + 6*l + 5), nmsg);
    node_state_kernel<<<NN / 16, 256, 0, stream>>>(
        hF, h16, nmsg,
        sW1[l], pw(OFF_STATE + 6*l + 1), sW2[l], pw(OFF_STATE + 6*l + 3),
        sW3[l], pw(OFF_STATE + 6*l + 5));
    node_scalar_mlp_kernel<<<NN / 16, 256, 0, stream>>>(
        h16, bW1[l], pw(OFF_BETA + 6*l + 1), bW2[l], pw(OFF_BETA + 6*l + 3),
        pw(OFF_BETA + 6*l + 4), pw(OFF_BETA + 6*l + 5), beta);
    node_scalar_mlp_kernel<<<NN / 16, 256, 0, stream>>>(
        h16, gW1[l], pw(OFF_GAMMA + 6*l + 1), gW2[l], pw(OFF_GAMMA + 6*l + 3),
        pw(OFF_GAMMA + 6*l + 4), pw(OFF_GAMMA + 6*l + 5), gmm);
    hipMemsetAsync(upd, 0, (size_t)NN * 3 * 4, stream);
    edge_alpha_kernel<<<EE / EPB, 256, 0, stream>>>(
        h16, xcur, xA, xB, bondA, bondB, src, dst,
        aW1[l], pw(OFF_ALPHA + 6*l + 1), aW2[l], pw(OFF_ALPHA + 6*l + 3),
        pw(OFF_ALPHA + 6*l + 4), pw(OFF_ALPHA + 6*l + 5), upd);
    x_update_kernel<<<NN / 256, 256, 0, stream>>>(xcur, upd, beta, gmm, s_in, xA, xB);
  }

  hipMemsetAsync(sums, 0, 16, stream);
  reduce_sum_kernel<<<NN / 256, 256, 0, stream>>>(xcur, sums);
  recenter_kernel<<<(NN * 3 + 255) / 256, 256, 0, stream>>>(xcur, sums, (float*)d_out);
}
